// GRUD_75462575390850
// MI455X (gfx1250) — compile-verified, matrix operand order checked by reading, not run on hardware
//
#include <hip/hip_runtime.h>

typedef __bf16 bf16_t;
typedef __attribute__((ext_vector_type(16))) __bf16 v16bf;
typedef __attribute__((ext_vector_type(8)))  __bf16 v8bf;
typedef __attribute__((ext_vector_type(8)))  float  v8f;

#define B_DIM 16384
#define E_DIM 1024
#define H_DIM 1024

// ---------- bf16 helpers (round-to-nearest-even) ----------
__device__ __forceinline__ unsigned short f2bf_u(float f) {
    unsigned u = __builtin_bit_cast(unsigned, f);
    unsigned r = 0x7FFFu + ((u >> 16) & 1u);
    return (unsigned short)((u + r) >> 16);
}
__device__ __forceinline__ float bf2f(unsigned short s) {
    unsigned u = ((unsigned)s) << 16;
    return __builtin_bit_cast(float, u);
}

// ---------- WMMA wrapper ----------
__device__ __forceinline__ v8f wmma_bf16(v16bf a, v16bf b, v8f c) {
    return __builtin_amdgcn_wmma_f32_16x16x32_bf16(false, a, false, b, (short)0, c, false, false);
}

// Fragment loader (CDNA5 ISA 7.12.2 16-bit A/B layout): laneBase = base + (row0 + r)*ld + hi*8
__device__ __forceinline__ v16bf ldfrag(const bf16_t* laneBase, int k0) {
    union U { v16bf f; v8bf h[2]; } u;
    u.h[0] = *(const v8bf*)(laneBase + k0);
    u.h[1] = *(const v8bf*)(laneBase + k0 + 16);
    return u.f;
}

// ---------- prep kernels ----------
__global__ void cvt_bf16_kernel(const float* __restrict__ src, unsigned short* __restrict__ dst, int n) {
    int i = blockIdx.x * blockDim.x + threadIdx.x;
    int stride = gridDim.x * blockDim.x;
    for (; i < n; i += stride) dst[i] = f2bf_u(src[i]);
}

// src [R,C] fp32 row-major -> dst [C,R] bf16 (transposed), LDS tiled
__global__ void transpose_bf16_kernel(const float* __restrict__ src, unsigned short* __restrict__ dst,
                                      int R, int C) {
    __shared__ unsigned short t[32][33];
    int r0 = blockIdx.y * 32, c0 = blockIdx.x * 32;
    int tx = threadIdx.x, ty = threadIdx.y;
    for (int i = ty; i < 32; i += 8)
        t[i][tx] = f2bf_u(src[(r0 + i) * C + (c0 + tx)]);
    __syncthreads();
    for (int i = ty; i < 32; i += 8)
        dst[(c0 + i) * R + (r0 + tx)] = t[tx][i];
}

// ---------- stage A: decay gates + masked blend + hidden decay ----------
// Per wave: 32(M) x 16(N) tile; 4 accumulators; B fragments shared across both M-subtiles.
__global__ void __launch_bounds__(128)
grud_decay_kernel(const float* __restrict__ x, const float* __restrict__ x_mask,
                  const float* __restrict__ x_last, const float* __restrict__ x_mean,
                  const float* __restrict__ hs,
                  const float* __restrict__ gx_b, const float* __restrict__ gh_b,
                  const bf16_t* __restrict__ delta_bf, const bf16_t* __restrict__ gxw_bf,
                  const bf16_t* __restrict__ ghw_bf,
                  unsigned short* __restrict__ xt_bf, unsigned short* __restrict__ h_bf) {
    const int lane = threadIdx.x & 31;
    const int wave = threadIdx.x >> 5;
    const int r    = lane & 15;
    const int hi   = lane >> 4;
    const int nBase = (blockIdx.x * 4 + wave) * 16;
    const int mBase = blockIdx.y * 32;

    const bf16_t* aBase0 = delta_bf + (mBase + r) * E_DIM + hi * 8;
    const bf16_t* aBase1 = aBase0 + 16 * E_DIM;
    const bf16_t* bxBase = gxw_bf + (nBase + r) * E_DIM + hi * 8;  // row j of gx_w == col j of gx_w.T
    const bf16_t* bhBase = ghw_bf + (nBase + r) * E_DIM + hi * 8;

    v8f accx0 = {}, accx1 = {}, acch0 = {}, acch1 = {};
    for (int k0 = 0; k0 < E_DIM; k0 += 32) {
        __builtin_prefetch(aBase0 + k0 + 256, 0, 1);
        v16bf a0 = ldfrag(aBase0, k0);
        v16bf a1 = ldfrag(aBase1, k0);
        v16bf bx = ldfrag(bxBase, k0);
        v16bf bh = ldfrag(bhBase, k0);
        accx0 = wmma_bf16(a0, bx, accx0);
        accx1 = wmma_bf16(a1, bx, accx1);
        acch0 = wmma_bf16(a0, bh, acch0);
        acch1 = wmma_bf16(a1, bh, acch1);
    }

    const int col = nBase + r;
    const float bxb = gx_b[col];
    const float bhb = gh_b[col];
#pragma unroll
    for (int t = 0; t < 2; ++t) {
        v8f ax = t ? accx1 : accx0;
        v8f ahh = t ? acch1 : acch0;
#pragma unroll
        for (int v = 0; v < 8; ++v) {
            int row = mBase + t * 16 + v + hi * 8;   // C/D layout: lanes 16-31 hold M = v+8
            int idx = row * E_DIM + col;
            float dx = __expf(-fmaxf(ax[v] + bxb, 0.0f));
            float dh = __expf(-fmaxf(ahh[v] + bhb, 0.0f));
            float m  = x_mask[idx];
            float xt = m * x[idx] + (1.0f - m) * (dx * x_last[idx] + (1.0f - dx) * x_mean[idx]);
            float h  = dh * hs[idx];
            xt_bf[idx] = f2bf_u(xt);
            h_bf[idx]  = f2bf_u(h);
        }
    }
}

// ---------- stage B: GRU gates GEMMs + fused epilogue ----------
// Per wave: 32(M) x 16(N) tile; 12 accumulators; 18 WMMAs per K-step from 15 fragment loads.
__global__ void __launch_bounds__(128)
grud_gates_kernel(const float* __restrict__ b_ih, const float* __restrict__ b_hh,
                  const bf16_t* __restrict__ xt_bf, const bf16_t* __restrict__ xm_bf,
                  const bf16_t* __restrict__ h_bf,
                  const bf16_t* __restrict__ wihT,  // [3H, 2E]
                  const bf16_t* __restrict__ whhT,  // [3H, H]
                  float* __restrict__ out) {
    const int lane = threadIdx.x & 31;
    const int wave = threadIdx.x >> 5;
    const int r    = lane & 15;
    const int hi   = lane >> 4;
    const int jBase = (blockIdx.x * 4 + wave) * 16;
    const int mBase = blockIdx.y * 32;

    const bf16_t* aXt0 = xt_bf + (mBase + r) * E_DIM + hi * 8;
    const bf16_t* aXt1 = aXt0 + 16 * E_DIM;
    const bf16_t* aXm0 = xm_bf + (mBase + r) * E_DIM + hi * 8;
    const bf16_t* aXm1 = aXm0 + 16 * E_DIM;
    const bf16_t* aH0  = h_bf + (mBase + r) * H_DIM + hi * 8;
    const bf16_t* aH1  = aH0 + 16 * H_DIM;

    const int ldw = 2 * E_DIM;
    const bf16_t* bR1 = wihT + (jBase +             r) * ldw + hi * 8;
    const bf16_t* bZ1 = wihT + (jBase +     H_DIM + r) * ldw + hi * 8;
    const bf16_t* bN1 = wihT + (jBase + 2 * H_DIM + r) * ldw + hi * 8;
    const bf16_t* bRh = whhT + (jBase +             r) * H_DIM + hi * 8;
    const bf16_t* bZh = whhT + (jBase +     H_DIM + r) * H_DIM + hi * 8;
    const bf16_t* bNh = whhT + (jBase + 2 * H_DIM + r) * H_DIM + hi * 8;

    v8f ir0 = {}, iz0 = {}, in0 = {}, hr0 = {}, hz0 = {}, hn0 = {};
    v8f ir1 = {}, iz1 = {}, in1 = {}, hr1 = {}, hz1 = {}, hn1 = {};

    for (int k0 = 0; k0 < E_DIM; k0 += 32) {
        __builtin_prefetch(aXt0 + k0 + 256, 0, 1);
        __builtin_prefetch(aH0  + k0 + 256, 0, 1);
        v16bf axt0 = ldfrag(aXt0, k0), axt1 = ldfrag(aXt1, k0);
        v16bf axm0 = ldfrag(aXm0, k0), axm1 = ldfrag(aXm1, k0);
        v16bf ah0  = ldfrag(aH0,  k0), ah1  = ldfrag(aH1,  k0);
        v16bf b;
        // input path: first E rows of w_ih hit x_t, next E rows hit x_mask
        b = ldfrag(bR1, k0);          ir0 = wmma_bf16(axt0, b, ir0); ir1 = wmma_bf16(axt1, b, ir1);
        b = ldfrag(bR1, k0 + E_DIM);  ir0 = wmma_bf16(axm0, b, ir0); ir1 = wmma_bf16(axm1, b, ir1);
        b = ldfrag(bZ1, k0);          iz0 = wmma_bf16(axt0, b, iz0); iz1 = wmma_bf16(axt1, b, iz1);
        b = ldfrag(bZ1, k0 + E_DIM);  iz0 = wmma_bf16(axm0, b, iz0); iz1 = wmma_bf16(axm1, b, iz1);
        b = ldfrag(bN1, k0);          in0 = wmma_bf16(axt0, b, in0); in1 = wmma_bf16(axt1, b, in1);
        b = ldfrag(bN1, k0 + E_DIM);  in0 = wmma_bf16(axm0, b, in0); in1 = wmma_bf16(axm1, b, in1);
        // hidden path
        b = ldfrag(bRh, k0);          hr0 = wmma_bf16(ah0, b, hr0);  hr1 = wmma_bf16(ah1, b, hr1);
        b = ldfrag(bZh, k0);          hz0 = wmma_bf16(ah0, b, hz0);  hz1 = wmma_bf16(ah1, b, hz1);
        b = ldfrag(bNh, k0);          hn0 = wmma_bf16(ah0, b, hn0);  hn1 = wmma_bf16(ah1, b, hn1);
    }

    const int col = jBase + r;
    const float bir = b_ih[col], biz = b_ih[col + H_DIM], bin = b_ih[col + 2 * H_DIM];
    const float bhr = b_hh[col], bhz = b_hh[col + H_DIM], bhn = b_hh[col + 2 * H_DIM];
    const unsigned short* h_us = (const unsigned short*)h_bf;
#pragma unroll
    for (int t = 0; t < 2; ++t) {
        v8f ir = t ? ir1 : ir0, iz = t ? iz1 : iz0, inv = t ? in1 : in0;
        v8f hr = t ? hr1 : hr0, hz = t ? hz1 : hz0, hnv = t ? hn1 : hn0;
#pragma unroll
        for (int v = 0; v < 8; ++v) {
            int row = mBase + t * 16 + v + hi * 8;
            int idx = row * H_DIM + col;
            float rg = 1.0f / (1.0f + __expf(-((ir[v] + bir) + (hr[v] + bhr))));
            float zg = 1.0f / (1.0f + __expf(-((iz[v] + biz) + (hz[v] + bhz))));
            float ng = tanhf((inv[v] + bin) + rg * (hnv[v] + bhn));
            float h  = bf2f(h_us[idx]);
            out[idx] = (1.0f - zg) * ng + zg * h;
        }
    }
}

extern "C" void kernel_launch(void* const* d_in, const int* in_sizes, int n_in,
                              void* d_out, int out_size, void* d_ws, size_t ws_size,
                              hipStream_t stream) {
    (void)in_sizes; (void)n_in; (void)out_size; (void)ws_size;
    const float* x      = (const float*)d_in[0];
    const float* x_mask = (const float*)d_in[1];
    const float* delta  = (const float*)d_in[2];
    const float* x_last = (const float*)d_in[3];
    const float* x_mean = (const float*)d_in[4];
    const float* hs     = (const float*)d_in[5];
    const float* gx_w   = (const float*)d_in[6];
    const float* gx_b   = (const float*)d_in[7];
    const float* gh_w   = (const float*)d_in[8];
    const float* gh_b   = (const float*)d_in[9];
    const float* w_ih   = (const float*)d_in[10];
    const float* b_ih   = (const float*)d_in[11];
    const float* w_hh   = (const float*)d_in[12];
    const float* b_hh   = (const float*)d_in[13];
    float* out = (float*)d_out;

    // ---- workspace layout (bf16 stored as unsigned short), 256B-aligned regions ----
    char* ws = (char*)d_ws;
    size_t off = 0;
    auto carve = [&](size_t bytes) -> char* {
        char* p = ws + off;
        off += (bytes + 255) & ~(size_t)255;
        return p;
    };
    unsigned short* delta_bf = (unsigned short*)carve((size_t)B_DIM * E_DIM * 2);   // 32 MB
    unsigned short* xmask_bf = (unsigned short*)carve((size_t)B_DIM * E_DIM * 2);   // 32 MB
    unsigned short* xt_bf    = (unsigned short*)carve((size_t)B_DIM * E_DIM * 2);   // 32 MB
    unsigned short* h_bf     = (unsigned short*)carve((size_t)B_DIM * H_DIM * 2);   // 32 MB
    unsigned short* gxw_bf   = (unsigned short*)carve((size_t)E_DIM * E_DIM * 2);   //  2 MB
    unsigned short* ghw_bf   = (unsigned short*)carve((size_t)E_DIM * E_DIM * 2);   //  2 MB
    unsigned short* wihT_bf  = (unsigned short*)carve((size_t)2 * E_DIM * 3 * H_DIM * 2); // 12 MB
    unsigned short* whhT_bf  = (unsigned short*)carve((size_t)H_DIM * 3 * H_DIM * 2);     //  6 MB

    // ---- stage 0: precision conversion / weight transposes ----
    cvt_bf16_kernel<<<4096, 256, 0, stream>>>(delta,  delta_bf, B_DIM * E_DIM);
    cvt_bf16_kernel<<<4096, 256, 0, stream>>>(x_mask, xmask_bf, B_DIM * E_DIM);
    cvt_bf16_kernel<<<1024, 256, 0, stream>>>(gx_w,   gxw_bf,   E_DIM * E_DIM);
    cvt_bf16_kernel<<<1024, 256, 0, stream>>>(gh_w,   ghw_bf,   E_DIM * E_DIM);
    transpose_bf16_kernel<<<dim3(3 * H_DIM / 32, 2 * E_DIM / 32), dim3(32, 8), 0, stream>>>(
        w_ih, wihT_bf, 2 * E_DIM, 3 * H_DIM);
    transpose_bf16_kernel<<<dim3(3 * H_DIM / 32, H_DIM / 32), dim3(32, 8), 0, stream>>>(
        w_hh, whhT_bf, H_DIM, 3 * H_DIM);

    // ---- stage A: decay GEMMs + blend (32x16 tile per wave, 4 waves/block) ----
    grud_decay_kernel<<<dim3(E_DIM / 64, B_DIM / 32), 128, 0, stream>>>(
        x, x_mask, x_last, x_mean, hs, gx_b, gh_b,
        (const bf16_t*)delta_bf, (const bf16_t*)gxw_bf, (const bf16_t*)ghw_bf,
        xt_bf, h_bf);

    // ---- stage B: gates GEMMs + GRU epilogue (32x16 tile per wave) ----
    grud_gates_kernel<<<dim3(H_DIM / 64, B_DIM / 32), 128, 0, stream>>>(
        b_ih, b_hh,
        (const bf16_t*)xt_bf, (const bf16_t*)xmask_bf, (const bf16_t*)h_bf,
        (const bf16_t*)wihT_bf, (const bf16_t*)whhT_bf,
        out);
}
